// Model_60284160967102
// MI455X (gfx1250) — matrix operand order repaired and checked by reading, not compiled, not run on hardware
//
#include <hip/hip_runtime.h>
#include <cstdint>
#include <cstddef>

// ---------------------------------------------------------------------------
// Mamba2 vision model forward for MI455X (gfx1250), bf16 WMMA everywhere.
// All matmuls lower to v_wmma_f32_16x16x32_bf16 (wave32, 16x16x32 tiles).
// - LDS B-operand tiles are K-pair-packed dwords: one ds_load_b32 per frag reg.
// - GEMM staging is branchless (clamped addresses) and double-buffered.
// - yintra builds decay-weighted A fragments directly from global float4 reads.
// Workspace requirement: ~397 MB carved from d_ws.
// ---------------------------------------------------------------------------

typedef __bf16 bf16;
typedef __attribute__((ext_vector_type(16))) __bf16 bf16x16;
typedef __attribute__((ext_vector_type(8)))  float  floatx8;
typedef __attribute__((ext_vector_type(8)))  unsigned int uint32x8;

union PairVec { uint32x8 u; bf16x16 v; };
union BF2     { uint32_t u; bf16 b[2]; };

#define D_MODEL   768
#define HEADDIM   64
#define D_STATE   64
#define NHEADS    24
#define D_INNER   1536
#define D_XBC     1664          // D_INNER + 2*D_STATE
#define D_IN_PROJ 3224          // 2*D_INNER + 2*D_STATE + NHEADS
#define T_SEQ     4162          // 1 + 64 + 1 + 4096
#define NVAL_     4096
#define BATCH     2
#define NTOK      (BATCH * T_SEQ)       // 8324
#define CHUNK_    256
#define NCHUNK    17
#define LPAD      (NCHUNK * CHUNK_)     // 4352
#define NBC       (BATCH * NCHUNK)      // 34

// --------------------------- WMMA helpers ----------------------------------

__device__ inline floatx8 wmma_bf16(const bf16x16& a, const bf16x16& b, floatx8 c) {
  return __builtin_amdgcn_wmma_f32_16x16x32_bf16(false, a, false, b, (short)0, c,
                                                 false, false);
}

// A-matrix (16x32 bf16) K striping per ISA 7.12.2:
// lanes 0-15: VGPR0..3 -> K 0..7, VGPR4..7 -> K 16..23 ; lanes 16-31: +8
__device__ inline int afrag_kbase(int v, int half) {
  return (v < 4 ? 2 * v : 16 + 2 * (v - 4)) + half * 8;
}

// A fragment from a row-major LDS tile of K-pair dwords: row stride lda_dw.
__device__ inline bf16x16 load_afrag_dw(const uint32_t* smem, int lda_dw, int m0) {
  int lane = threadIdx.x & 31;
  int m = m0 + (lane & 15);
  int half = lane >> 4;
  const uint32_t* r = smem + m * lda_dw + 4 * half;  // kb/2 = (v<4?v:4+v) + 4*half
  PairVec f;
#pragma unroll
  for (int v = 0; v < 8; ++v) f.u[v] = r[(v < 4) ? v : 4 + v];
  return f.v;
}

// B fragment from K-pair-packed LDS tile: smem[kpair][n], row stride ldb_dw.
// VGPR v holds K = 2*(v<4?v:4+v) + 8*half -> kpair row = (v<4?v:4+v) + 4*half.
__device__ inline bf16x16 load_bfrag_dw(const uint32_t* smem, int ldb_dw, int n0) {
  int lane = threadIdx.x & 31;
  int n = n0 + (lane & 15);
  int half = lane >> 4;
  const uint32_t* cbase = smem + (4 * half) * ldb_dw + n;
  PairVec f;
#pragma unroll
  for (int v = 0; v < 8; ++v) f.u[v] = cbase[((v < 4) ? v : 4 + v) * ldb_dw];
  return f.v;
}

__device__ inline int perm_t(int t) {
  if (t < 66) return t;
  int p = t - 66;
  int r = p >> 6, c = p & 63;
  return 66 + (4095 - (c * 64 + r));
}

// ---------------------- generic bf16 WMMA GEMM -----------------------------
// C[M,N] = A[M,K] @ B[K,N] (+ resid), A,B row-major bf16, C f32.
// Requires: K % 32 == 0, N % 8 == 0 (holds for 768/1536/3224).
// OOB staging is handled by clamping addresses: garbage only ever reaches
// output elements that the (guarded) epilogue never stores.
#define GBM 128
#define GBN 128
#define GBK 32

__global__ __launch_bounds__(256) void gemm_bf16_rr(
    const bf16* __restrict__ A, const bf16* __restrict__ B,
    float* __restrict__ C, const float* __restrict__ resid,
    int M, int N, int K) {
  __shared__ uint32_t As[2][GBM][GBK / 2 + 1];   // [buf][m][kpair]
  __shared__ uint32_t Bs[2][GBK / 2][GBN];       // [buf][kpair][n]
  const uint16_t* Au = (const uint16_t*)A;
  const uint16_t* Bu = (const uint16_t*)B;
  int m0 = blockIdx.y * GBM, n0 = blockIdx.x * GBN;
  int tid = threadIdx.x;
  int w = tid >> 5, lane = tid & 31;
  int wm = (w & 3) * 32;
  int wn = (w >> 2) * 64;
  floatx8 acc[2][4];
#pragma unroll
  for (int i = 0; i < 2; ++i)
#pragma unroll
    for (int j = 0; j < 4; ++j) acc[i][j] = (floatx8)(0.0f);

  int ar = tid >> 1, acq = (tid & 1) * 8;     // A: row, dword col
  int kr = tid >> 4, bn0 = (tid & 15) * 8;    // B: k-pair row, 8 cols
  int gmA = m0 + ar;
  if (gmA >= M) gmA = M - 1;                  // clamp (see header comment)
  int bcol = n0 + bn0;
  if (bcol > N - 8) bcol = N - 8;             // clamp; N%8==0 -> no straddle
  const uint32_t* arow = (const uint32_t*)(Au + (size_t)gmA * K);
  const uint16_t* bqb = Bu + bcol;

  auto stage = [&](int buf, int k0) {
    const uint32_t* src = arow + k0 / 2 + acq;
    uint4 v0 = ((const uint4*)src)[0];
    uint4 v1 = ((const uint4*)src)[1];
    uint32_t* ad = &As[buf][ar][acq];
    ad[0] = v0.x; ad[1] = v0.y; ad[2] = v0.z; ad[3] = v0.w;
    ad[4] = v1.x; ad[5] = v1.y; ad[6] = v1.z; ad[7] = v1.w;
    const uint16_t* b0 = bqb + (size_t)(k0 + 2 * kr) * N;
    uint4 lo = *(const uint4*)b0;
    uint4 hi = *(const uint4*)(b0 + N);
    uint32_t* bd = &Bs[buf][kr][bn0];
    bd[0] = (lo.x & 0xffffu) | (hi.x << 16);
    bd[1] = (lo.x >> 16)     | (hi.x & 0xffff0000u);
    bd[2] = (lo.y & 0xffffu) | (hi.y << 16);
    bd[3] = (lo.y >> 16)     | (hi.y & 0xffff0000u);
    bd[4] = (lo.z & 0xffffu) | (hi.z << 16);
    bd[5] = (lo.z >> 16)     | (hi.z & 0xffff0000u);
    bd[6] = (lo.w & 0xffffu) | (hi.w << 16);
    bd[7] = (lo.w >> 16)     | (hi.w & 0xffff0000u);
  };

  stage(0, 0);
  int cur = 0;
  for (int k0 = 0; k0 < K; k0 += GBK) {
    __syncthreads();
    if (k0 + GBK < K) stage(cur ^ 1, k0 + GBK);
    if (k0 + 2 * GBK < K)
      __builtin_prefetch(arow + (k0 + 2 * GBK) / 2, 0, 1);
    bf16x16 af0 = load_afrag_dw(&As[cur][0][0], GBK / 2 + 1, wm);
    bf16x16 af1 = load_afrag_dw(&As[cur][0][0], GBK / 2 + 1, wm + 16);
#pragma unroll
    for (int ni = 0; ni < 4; ++ni) {
      bf16x16 bfr = load_bfrag_dw(&Bs[cur][0][0], GBN, wn + ni * 16);
      acc[0][ni] = wmma_bf16(af0, bfr, acc[0][ni]);
      acc[1][ni] = wmma_bf16(af1, bfr, acc[1][ni]);
    }
    cur ^= 1;
  }
  int half = lane >> 4, nl = lane & 15;
  bool interior = (m0 + GBM <= M) && (n0 + GBN <= N);
  if (interior) {
#pragma unroll
    for (int mi = 0; mi < 2; ++mi)
#pragma unroll
      for (int ni = 0; ni < 4; ++ni)
#pragma unroll
        for (int r = 0; r < 8; ++r) {
          int gm = m0 + wm + mi * 16 + r + 8 * half;
          int gn = n0 + wn + ni * 16 + nl;
          float v = acc[mi][ni][r];
          if (resid) v += resid[(size_t)gm * N + gn];
          C[(size_t)gm * N + gn] = v;
        }
  } else {
#pragma unroll
    for (int mi = 0; mi < 2; ++mi)
#pragma unroll
      for (int ni = 0; ni < 4; ++ni)
#pragma unroll
        for (int r = 0; r < 8; ++r) {
          int gm = m0 + wm + mi * 16 + r + 8 * half;
          int gn = n0 + wn + ni * 16 + nl;
          if (gm < M && gn < N) {
            float v = acc[mi][ni][r];
            if (resid) v += resid[(size_t)gm * N + gn];
            C[(size_t)gm * N + gn] = v;
          }
        }
  }
}

// ----------------------- SSD: scores = C @ B^T -----------------------------
// per (b,chunk): scores[t,s] = sum_n Cc[t,n] * Bc[s,n]; M=N=256, K=64.
// Both K-slices staged up front -> single barrier, 16 WMMAs/wave after it.
__global__ __launch_bounds__(256) void ssd_scores_k(
    const bf16* __restrict__ Cc, const bf16* __restrict__ Bc,
    float* __restrict__ scores) {
  __shared__ uint32_t As[128][33];   // kpairs 0..31 (K=0..63), pad 1
  __shared__ uint32_t Bs[32][128];   // [kpair][s]
  int bc = blockIdx.z;
  int base = (bc / NCHUNK) * LPAD + (bc % NCHUNK) * CHUNK_;
  int m0 = blockIdx.y * 128, n0 = blockIdx.x * 128;
  int tid = threadIdx.x;
  int w = tid >> 5, lane = tid & 31;
  int wm = (w & 3) * 32, wn = (w >> 2) * 64;
  floatx8 acc[2][4];
#pragma unroll
  for (int i = 0; i < 2; ++i)
#pragma unroll
    for (int j = 0; j < 4; ++j) acc[i][j] = (floatx8)(0.0f);

  const uint32_t* Cdw = (const uint32_t*)Cc;
  const uint32_t* Bdw = (const uint32_t*)Bc;
  {
    int ar = tid >> 1, acq = (tid & 1) * 16;
    const uint32_t* src = Cdw + (size_t)(base + m0 + ar) * 32 + acq;
#pragma unroll
    for (int q = 0; q < 4; ++q) {
      uint4 v = ((const uint4*)src)[q];
      uint32_t* ad = &As[ar][acq + 4 * q];
      ad[0] = v.x; ad[1] = v.y; ad[2] = v.z; ad[3] = v.w;
    }
    int kr = tid >> 4, sn0 = (tid & 15) * 8;
#pragma unroll
    for (int half = 0; half < 2; ++half)     // k0 = 0, 32
#pragma unroll
      for (int j = 0; j < 8; ++j)
        Bs[kr + 16 * half][sn0 + j] =
            Bdw[(size_t)(base + n0 + sn0 + j) * 32 + 16 * half + kr];
  }
  __syncthreads();
#pragma unroll
  for (int k0 = 0; k0 < 64; k0 += 32) {
    bf16x16 af0 = load_afrag_dw(&As[0][k0 / 2], 33, wm);
    bf16x16 af1 = load_afrag_dw(&As[0][k0 / 2], 33, wm + 16);
#pragma unroll
    for (int ni = 0; ni < 4; ++ni) {
      bf16x16 bfr = load_bfrag_dw(&Bs[k0 / 2][0], 128, wn + ni * 16);
      acc[0][ni] = wmma_bf16(af0, bfr, acc[0][ni]);
      acc[1][ni] = wmma_bf16(af1, bfr, acc[1][ni]);
    }
  }
  int half = lane >> 4, nl = lane & 15;
#pragma unroll
  for (int mi = 0; mi < 2; ++mi)
#pragma unroll
    for (int ni = 0; ni < 4; ++ni)
#pragma unroll
      for (int r = 0; r < 8; ++r) {
        int gm = m0 + wm + mi * 16 + r + 8 * half;
        int gn = n0 + wn + ni * 16 + nl;
        scores[(size_t)bc * 65536 + gm * 256 + gn] = acc[mi][ni][r];
      }
}

// --------------- SSD: intra-chunk y = (scores*decay*dt) @ X ----------------
// grid (NHEADS, NBC). W[t,s] fragments are built straight from global float4
// reads of the scores tile (each element read exactly once per workgroup);
// the small X tile is double-buffered in LDS.
__global__ __launch_bounds__(256) void ssd_yintra_k(
    const float* __restrict__ scores, const bf16* __restrict__ xs,
    const float* __restrict__ cs, const float* __restrict__ dt,
    float* __restrict__ ybuf) {
  __shared__ uint32_t Xt2[2][16][64];     // X tile, K-pair-packed [spair][p]
  __shared__ float cs_sh[CHUNK_], dt_sh[CHUNK_];
  int h = blockIdx.x, bc = blockIdx.y;
  int b = bc / NCHUNK, c = bc % NCHUNK;
  int tbase = b * LPAD + c * CHUNK_;
  int tid = threadIdx.x;
  cs_sh[tid] = cs[(size_t)(tbase + tid) * NHEADS + h];
  dt_sh[tid] = dt[(size_t)(tbase + tid) * NHEADS + h];
  int w = tid >> 5, lane = tid & 31;
  int trow = w * 32;
  const uint16_t* xsu = (const uint16_t*)xs;
  floatx8 acc[2][4];
#pragma unroll
  for (int i = 0; i < 2; ++i)
#pragma unroll
    for (int j = 0; j < 4; ++j) acc[i][j] = (floatx8)(0.0f);

  int kr = tid >> 4, p0 = (tid & 15) * 4;
  auto stageX = [&](int buf, int s0) {
    const uint16_t* x0 = xsu + (size_t)(tbase + s0 + 2 * kr) * D_INNER + h * 64 + p0;
    uint2 lo2 = *(const uint2*)x0;
    uint2 hi2 = *(const uint2*)(x0 + D_INNER);
    uint32_t* xd = &Xt2[buf][kr][p0];
    xd[0] = (lo2.x & 0xffffu) | (hi2.x << 16);
    xd[1] = (lo2.x >> 16)     | (hi2.x & 0xffff0000u);
    xd[2] = (lo2.y & 0xffffu) | (hi2.y << 16);
    xd[3] = (lo2.y >> 16)     | (hi2.y & 0xffff0000u);
  };

  stageX(0, 0);
  __syncthreads();
  int cur = 0;
  int mbase = trow + (lane & 15);
  int halfk = lane >> 4;
  for (int s0 = 0; s0 < CHUNK_; s0 += 32) {
    if (s0 + 32 < CHUNK_) stageX(cur ^ 1, s0 + 32);
    bf16x16 afr[2];
#pragma unroll
    for (int mi = 0; mi < 2; ++mi) {
      int m = mbase + mi * 16;
      const float* srow = scores + (size_t)bc * 65536 + (size_t)m * 256 + s0 + 8 * halfk;
      float4 fa = ((const float4*)srow)[0];
      float4 fb = ((const float4*)srow)[1];
      float4 fc = ((const float4*)(srow + 16))[0];
      float4 fd = ((const float4*)(srow + 16))[1];
      float sv[16] = {fa.x, fa.y, fa.z, fa.w, fb.x, fb.y, fb.z, fb.w,
                      fc.x, fc.y, fc.z, fc.w, fd.x, fd.y, fd.z, fd.w};
      bf16x16 af;
#pragma unroll
      for (int v = 0; v < 8; ++v) {
        int kb = afrag_kbase(v, halfk);
#pragma unroll
        for (int j = 0; j < 2; ++j) {
          int idx = (v < 4) ? (2 * v + j) : (8 + 2 * (v - 4) + j);
          int s = s0 + kb + j;
          float d = (m >= s) ? __expf(cs_sh[m] - cs_sh[s]) : 0.0f;
          af[2 * v + j] = (bf16)(sv[idx] * d * dt_sh[s]);
        }
      }
      afr[mi] = af;
    }
#pragma unroll
    for (int ni = 0; ni < 4; ++ni) {
      bf16x16 bfr = load_bfrag_dw(&Xt2[cur][0][0], 64, ni * 16);
      acc[0][ni] = wmma_bf16(afr[0], bfr, acc[0][ni]);
      acc[1][ni] = wmma_bf16(afr[1], bfr, acc[1][ni]);
    }
    __syncthreads();
    cur ^= 1;
  }
  int half = lane >> 4, nl = lane & 15;
#pragma unroll
  for (int mi = 0; mi < 2; ++mi)
#pragma unroll
    for (int ni = 0; ni < 4; ++ni)
#pragma unroll
      for (int r = 0; r < 8; ++r) {
        int gm = trow + mi * 16 + r + 8 * half;
        int gn = ni * 16 + nl;
        ybuf[(size_t)(tbase + gm) * D_INNER + h * 64 + gn] = acc[mi][ni][r];
      }
}

// -------------- SSD: chunk state S[p,n] = sum_s G[s]*x[s,p]*B[s,n] ---------
__global__ __launch_bounds__(256) void ssd_S_k(
    const bf16* __restrict__ Bc, const bf16* __restrict__ xs,
    const float* __restrict__ cs, const float* __restrict__ dt,
    float* __restrict__ Sbuf) {
  __shared__ bf16 Xt[32][64 + 2];
  __shared__ uint32_t Bt2[16][64];        // B tile, K-pair-packed [spair][n]
  __shared__ float G_sh[32];
  __shared__ float cs_sh[CHUNK_], dt_sh[CHUNK_];
  int h = blockIdx.x, bc = blockIdx.y;
  int b = bc / NCHUNK, c = bc % NCHUNK;
  int tbase = b * LPAD + c * CHUNK_;
  int tid = threadIdx.x;
  cs_sh[tid] = cs[(size_t)(tbase + tid) * NHEADS + h];
  dt_sh[tid] = dt[(size_t)(tbase + tid) * NHEADS + h];
  int w = tid >> 5, lane = tid & 31;
  int mi = w >> 1;            // 4 m-tiles of 16 (p)
  int njb = (w & 1) * 2;      // 2 n-tiles each
  const uint16_t* xsu = (const uint16_t*)xs;
  const uint16_t* Bu = (const uint16_t*)Bc;
  floatx8 acc[2];
  acc[0] = (floatx8)(0.0f); acc[1] = (floatx8)(0.0f);
  __syncthreads();
  float cslast = cs_sh[CHUNK_ - 1];

  int xr = tid >> 3, xc0 = (tid & 7) * 8;
  int kr = tid >> 4, p0 = (tid & 15) * 4;
  uint32_t* Xtd = (uint32_t*)&Xt[0][0];
  for (int s0 = 0; s0 < CHUNK_; s0 += 32) {
    {
      const uint16_t* x0 = xsu + (size_t)(tbase + s0 + xr) * D_INNER + h * 64 + xc0;
      uint4 v = *(const uint4*)x0;
      uint32_t* dst = Xtd + xr * 33 + xc0 / 2;
      dst[0] = v.x; dst[1] = v.y; dst[2] = v.z; dst[3] = v.w;
    }
    {
      const uint16_t* b0 = Bu + (size_t)(tbase + s0 + 2 * kr) * 64 + p0;
      uint2 lo2 = *(const uint2*)b0;
      uint2 hi2 = *(const uint2*)(b0 + 64);
      Bt2[kr][p0 + 0] = (lo2.x & 0xffffu) | (hi2.x << 16);
      Bt2[kr][p0 + 1] = (lo2.x >> 16)     | (hi2.x & 0xffff0000u);
      Bt2[kr][p0 + 2] = (lo2.y & 0xffffu) | (hi2.y << 16);
      Bt2[kr][p0 + 3] = (lo2.y >> 16)     | (hi2.y & 0xffff0000u);
    }
    if (tid < 32)
      G_sh[tid] = __expf(cslast - cs_sh[s0 + tid]) * dt_sh[s0 + tid];
    __syncthreads();

    int halfk = lane >> 4;
    int m = mi * 16 + (lane & 15);    // p index
    bf16x16 af;
#pragma unroll
    for (int v = 0; v < 8; ++v) {
      int kb = afrag_kbase(v, halfk);
#pragma unroll
      for (int j = 0; j < 2; ++j) {
        int sl = kb + j;
        af[2 * v + j] = (bf16)(G_sh[sl] * (float)Xt[sl][m]);
      }
    }
#pragma unroll
    for (int jj = 0; jj < 2; ++jj) {
      bf16x16 bfr = load_bfrag_dw(&Bt2[0][0], 64, (njb + jj) * 16);
      acc[jj] = wmma_bf16(af, bfr, acc[jj]);
    }
    __syncthreads();
  }
  int half = lane >> 4, nl = lane & 15;
#pragma unroll
  for (int jj = 0; jj < 2; ++jj)
#pragma unroll
    for (int r = 0; r < 8; ++r) {
      int gm = mi * 16 + r + 8 * half;
      int gn = (njb + jj) * 16 + nl;
      Sbuf[((size_t)(bc * NHEADS + h) * 64 + gm) * 64 + gn] = acc[jj][r];
    }
}

// -------------- SSD: sequential inter-chunk scan over 17 chunks ------------
__global__ __launch_bounds__(256) void ssd_scan_k(
    const float* __restrict__ Sbuf, const float* __restrict__ cs,
    float* __restrict__ prevbuf) {
  int bh = blockIdx.x;
  int b = bh / NHEADS, h = bh % NHEADS;
  float st[16];
#pragma unroll
  for (int j = 0; j < 16; ++j) st[j] = 0.0f;
  for (int c = 0; c < NCHUNK; ++c) {
    float cdec = __expf(cs[(size_t)(b * LPAD + c * CHUNK_ + CHUNK_ - 1) * NHEADS + h]);
    int bc = b * NCHUNK + c;
    size_t base = (size_t)(bc * NHEADS + h) * 4096;
#pragma unroll
    for (int j = 0; j < 16; ++j) {
      int e = threadIdx.x + j * 256;
      prevbuf[base + e] = st[j];
      st[j] = st[j] * cdec + Sbuf[base + e];
    }
  }
}

// -------- SSD: y += exp(cs[t]) * C @ prev^T + D*x  (final combine) ---------
__global__ __launch_bounds__(256) void ssd_yinter_k(
    const bf16* __restrict__ Cc, const float* __restrict__ prev,
    const float* __restrict__ cs, const bf16* __restrict__ xs,
    const float* __restrict__ Dp, float* __restrict__ ybuf) {
  __shared__ bf16 Ct[CHUNK_][64 + 2];     // A tile, row-major (pairs adjacent)
  __shared__ uint32_t Pt2[32][64];        // prev^T K-pair-packed [npair][p]
  __shared__ float cs_sh[CHUNK_];
  int h = blockIdx.x, bc = blockIdx.y;
  int b = bc / NCHUNK, c = bc % NCHUNK;
  int tbase = b * LPAD + c * CHUNK_;
  int tid = threadIdx.x;
  cs_sh[tid] = cs[(size_t)(tbase + tid) * NHEADS + h];
  {
    const uint16_t* crow = (const uint16_t*)Cc + (size_t)(tbase + tid) * 64;
    uint32_t* Ctd = (uint32_t*)&Ct[0][0] + tid * 33;
#pragma unroll
    for (int j = 0; j < 8; ++j) {
      uint4 v = ((const uint4*)crow)[j];
      Ctd[4 * j + 0] = v.x; Ctd[4 * j + 1] = v.y;
      Ctd[4 * j + 2] = v.z; Ctd[4 * j + 3] = v.w;
    }
  }
  {
    int nh = tid >> 3, p0 = (tid & 7) * 8;
    const float* pb = prev + ((size_t)(bc * NHEADS + h) * 64) * 64 + 2 * nh;
#pragma unroll
    for (int j = 0; j < 8; ++j) {
      int p = p0 + j;
      float2 pv = *(const float2*)(pb + (size_t)p * 64);
      BF2 t; t.b[0] = (bf16)pv.x; t.b[1] = (bf16)pv.y;
      Pt2[nh][p] = t.u;
    }
  }
  int w = tid >> 5, lane = tid & 31;
  int trow = w * 32;
  floatx8 acc[2][4];
#pragma unroll
  for (int i = 0; i < 2; ++i)
#pragma unroll
    for (int j = 0; j < 4; ++j) acc[i][j] = (floatx8)(0.0f);
  __syncthreads();

  const uint32_t* Ctd = (const uint32_t*)&Ct[0][0];
#pragma unroll
  for (int k0 = 0; k0 < 64; k0 += 32) {
    bf16x16 af0 = load_afrag_dw(Ctd + k0 / 2, 33, trow);
    bf16x16 af1 = load_afrag_dw(Ctd + k0 / 2, 33, trow + 16);
#pragma unroll
    for (int ni = 0; ni < 4; ++ni) {
      bf16x16 bfr = load_bfrag_dw(&Pt2[0][0] + (k0 / 2) * 64, 64, ni * 16);
      acc[0][ni] = wmma_bf16(af0, bfr, acc[0][ni]);
      acc[1][ni] = wmma_bf16(af1, bfr, acc[1][ni]);
    }
  }
  float Dh = Dp[h];
  int half = lane >> 4, nl = lane & 15;
#pragma unroll
  for (int mi = 0; mi < 2; ++mi)
#pragma unroll
    for (int ni = 0; ni < 4; ++ni)
#pragma unroll
      for (int r = 0; r < 8; ++r) {
        int gm = trow + mi * 16 + r + 8 * half;
        int gn = ni * 16 + nl;
        size_t yi = (size_t)(tbase + gm) * D_INNER + h * 64 + gn;
        float v = ybuf[yi] + __expf(cs_sh[gm]) * acc[mi][ni][r] +
                  Dh * (float)xs[yi];
        ybuf[yi] = v;
      }
}

// --------------------------- elementwise glue ------------------------------

__global__ __launch_bounds__(256) void layernorm_bf16_k(
    const float* __restrict__ x, const float* __restrict__ g,
    const float* __restrict__ bta, bf16* __restrict__ out) {
  __shared__ float red[256];
  int tok = blockIdx.x, tid = threadIdx.x;
  const float* xp = x + (size_t)tok * D_MODEL;
  float v[3];
#pragma unroll
  for (int j = 0; j < 3; ++j) v[j] = xp[tid + j * 256];
  red[tid] = v[0] + v[1] + v[2];
  __syncthreads();
  for (int off = 128; off > 0; off >>= 1) {
    if (tid < off) red[tid] += red[tid + off];
    __syncthreads();
  }
  float mean = red[0] * (1.0f / 768.0f);
  __syncthreads();
  float sq = 0.0f;
#pragma unroll
  for (int j = 0; j < 3; ++j) { v[j] -= mean; sq += v[j] * v[j]; }
  red[tid] = sq;
  __syncthreads();
  for (int off = 128; off > 0; off >>= 1) {
    if (tid < off) red[tid] += red[tid + off];
    __syncthreads();
  }
  float rinv = rsqrtf(red[0] * (1.0f / 768.0f) + 1e-5f);
#pragma unroll
  for (int j = 0; j < 3; ++j) {
    int i = tid + j * 256;
    out[(size_t)tok * D_MODEL + i] = (bf16)(v[j] * rinv * g[i] + bta[i]);
  }
}

__global__ __launch_bounds__(256) void dtsoft_k(
    const float* __restrict__ zx, const float* __restrict__ dt_bias,
    float* __restrict__ dtp) {
  int idx = blockIdx.x * 256 + threadIdx.x;
  if (idx >= BATCH * LPAD * NHEADS) return;
  int h = idx % NHEADS;
  int tp = (idx / NHEADS) % LPAD;
  int b = idx / (NHEADS * LPAD);
  float v = 0.0f;
  if (tp < T_SEQ) {
    float xv = zx[(size_t)(b * T_SEQ + tp) * D_IN_PROJ + (D_IN_PROJ - NHEADS) + h] +
               dt_bias[h];
    v = (xv > 20.0f) ? xv : log1pf(__expf(xv));
  }
  dtp[idx] = v;
}

__global__ __launch_bounds__(256) void convsilu_k(
    const float* __restrict__ zx, const float* __restrict__ cw,
    const float* __restrict__ cb, bf16* __restrict__ xs,
    bf16* __restrict__ Bc, bf16* __restrict__ Cc) {
  int idx = blockIdx.x * 256 + threadIdx.x;
  if (idx >= BATCH * LPAD * D_XBC) return;
  int ch = idx % D_XBC;
  int tp = (idx / D_XBC) % LPAD;
  int b = idx / (D_XBC * LPAD);
  float v = 0.0f;
  if (tp < T_SEQ) {
    float a = cb[ch];
#pragma unroll
    for (int k = 0; k < 4; ++k) {
      int ts = tp - 3 + k;
      if (ts >= 0)
        a += zx[(size_t)(b * T_SEQ + ts) * D_IN_PROJ + D_INNER + ch] * cw[ch * 4 + k];
    }
    v = a / (1.0f + __expf(-a));
  }
  size_t row = (size_t)b * LPAD + tp;
  if (ch < D_INNER)                xs[row * D_INNER + ch] = (bf16)v;
  else if (ch < D_INNER + D_STATE) Bc[row * 64 + (ch - D_INNER)] = (bf16)v;
  else                             Cc[row * 64 + (ch - D_INNER - D_STATE)] = (bf16)v;
}

__global__ __launch_bounds__(256) void cs_k(
    const float* __restrict__ dtp, const float* __restrict__ A_log,
    float* __restrict__ csp) {
  int id = blockIdx.x * 256 + threadIdx.x;
  if (id >= NBC * NHEADS) return;
  int h = id % NHEADS;
  int bc = id / NHEADS;
  int b = bc / NCHUNK, c = bc % NCHUNK;
  float A = -__expf(A_log[h]);
  float run = 0.0f;
  size_t base = (size_t)(b * LPAD + c * CHUNK_) * NHEADS + h;
  for (int t = 0; t < CHUNK_; ++t) {
    run += dtp[base + (size_t)t * NHEADS] * A;
    csp[base + (size_t)t * NHEADS] = run;
  }
}

__global__ __launch_bounds__(256) void gaterms_k(
    const float* __restrict__ ybuf, const float* __restrict__ zx,
    const float* __restrict__ nw, bf16* __restrict__ out) {
  __shared__ float red[256];
  int tok = blockIdx.x;
  int b = tok / T_SEQ, t = tok % T_SEQ;
  const float* yp = ybuf + (size_t)(b * LPAD + t) * D_INNER;
  const float* zp = zx + (size_t)tok * D_IN_PROJ;
  int tid = threadIdx.x;
  float gv[6];
  float sq = 0.0f;
#pragma unroll
  for (int j = 0; j < 6; ++j) {
    int i = tid + j * 256;
    float z = zp[i];
    float g = yp[i] * z / (1.0f + __expf(-z));
    gv[j] = g;
    sq += g * g;
  }
  red[tid] = sq;
  __syncthreads();
  for (int off = 128; off > 0; off >>= 1) {
    if (tid < off) red[tid] += red[tid + off];
    __syncthreads();
  }
  float scale = rsqrtf(red[0] * (1.0f / 1536.0f) + 1e-5f);
#pragma unroll
  for (int j = 0; j < 6; ++j) {
    int i = tid + j * 256;
    out[(size_t)tok * D_INNER + i] = (bf16)(gv[j] * scale * nw[i]);
  }
}

__global__ __launch_bounds__(256) void perm_k(const float* __restrict__ X,
                                              float* __restrict__ Xs) {
  int idx = blockIdx.x * 256 + threadIdx.x;
  if (idx >= NTOK * D_MODEL) return;
  int d = idx % D_MODEL;
  int tok = idx / D_MODEL;
  int b = tok / T_SEQ, t = tok % T_SEQ;
  Xs[idx] = X[((size_t)b * T_SEQ + perm_t(t)) * D_MODEL + d];
}

__global__ __launch_bounds__(256) void combine_k(const float* __restrict__ A,
                                                 const float* __restrict__ B,
                                                 float* __restrict__ X) {
  int idx = blockIdx.x * 256 + threadIdx.x;
  if (idx >= NTOK * D_MODEL) return;
  int d = idx % D_MODEL;
  int tok = idx / D_MODEL;
  int b = tok / T_SEQ, t = tok % T_SEQ;
  X[idx] = A[idx] + B[((size_t)b * T_SEQ + perm_t(t)) * D_MODEL + d];
}

__global__ __launch_bounds__(256) void embed_k(
    const float* __restrict__ im8, const float* __restrict__ im64b,
    const float* __restrict__ frw, const float* __restrict__ frb,
    const float* __restrict__ s0, const float* __restrict__ s1,
    const float* __restrict__ suffix, float* __restrict__ X) {
  int idx = blockIdx.x * 256 + threadIdx.x;
  if (idx >= NTOK * D_MODEL) return;
  int d = idx % D_MODEL;
  int tok = idx / D_MODEL;
  int b = tok / T_SEQ, t = tok % T_SEQ;
  float v;
  if (t == 0) {
    v = s0[d];
  } else if (t <= 64) {
    const float* px = im8 + ((size_t)b * 64 + (t - 1)) * 3;
    v = px[0] * frw[d] + px[1] * frw[D_MODEL + d] + px[2] * frw[2 * D_MODEL + d] + frb[d];
  } else if (t == 65) {
    v = s1[d];
  } else {
    int pp = t - 66;
    const float* px = im64b + ((size_t)b * NVAL_ + pp) * 3;
    v = px[0] * frw[d] + px[1] * frw[D_MODEL + d] + px[2] * frw[2 * D_MODEL + d] +
        frb[d] + suffix[(size_t)pp * D_MODEL + d];
  }
  X[idx] = v;
}

__global__ __launch_bounds__(256) void resize_k(const float* __restrict__ im8,
                                                float* __restrict__ im64b) {
  int idx = blockIdx.x * 256 + threadIdx.x;
  if (idx >= BATCH * NVAL_ * 3) return;
  int c = idx % 3;
  int p = (idx / 3) % NVAL_;
  int b = idx / (3 * NVAL_);
  int r = p >> 6, q = p & 63;
  float sy = fminf(fmaxf((r + 0.5f) * 0.125f - 0.5f, 0.0f), 7.0f);
  float sx = fminf(fmaxf((q + 0.5f) * 0.125f - 0.5f, 0.0f), 7.0f);
  int y0 = (int)sy, x0 = (int)sx;
  int y1 = y0 + 1 > 7 ? 7 : y0 + 1;
  int x1 = x0 + 1 > 7 ? 7 : x0 + 1;
  float fy = sy - (float)y0, fx = sx - (float)x0;
  auto g = [&](int yy, int xx) {
    return im8[(((size_t)b * 8 + yy) * 8 + xx) * 3 + c];
  };
  im64b[idx] = (1.0f - fy) * ((1.0f - fx) * g(y0, x0) + fx * g(y0, x1)) +
               fy * ((1.0f - fx) * g(y1, x0) + fx * g(y1, x1));
}

__global__ __launch_bounds__(256) void torgb_k(
    const float* __restrict__ X, const float* __restrict__ w,
    const float* __restrict__ bias, float* __restrict__ im64b) {
  __shared__ float sh[3][256];
  int bp = blockIdx.x;
  int b = bp / NVAL_, p = bp % NVAL_;
  const float* xp = X + ((size_t)b * T_SEQ + 66 + p) * D_MODEL;
  int tid = threadIdx.x;
  float a0 = 0.0f, a1 = 0.0f, a2 = 0.0f;
  for (int d = tid; d < D_MODEL; d += 256) {
    float xv = xp[d];
    a0 += xv * w[d * 3 + 0];
    a1 += xv * w[d * 3 + 1];
    a2 += xv * w[d * 3 + 2];
  }
  sh[0][tid] = a0; sh[1][tid] = a1; sh[2][tid] = a2;
  __syncthreads();
  for (int off = 128; off > 0; off >>= 1) {
    if (tid < off) {
      sh[0][tid] += sh[0][tid + off];
      sh[1][tid] += sh[1][tid + off];
      sh[2][tid] += sh[2][tid + off];
    }
    __syncthreads();
  }
  if (tid < 3) im64b[(size_t)bp * 3 + tid] = sh[tid][0] + bias[tid];
}

__global__ __launch_bounds__(256) void cvt_bf16_k(const float* __restrict__ in,
                                                  bf16* __restrict__ out, int n) {
  int idx = blockIdx.x * 256 + threadIdx.x;
  if (idx < n) out[idx] = (bf16)in[idx];
}

__global__ __launch_bounds__(256) void predcopy_k(const float* __restrict__ src,
                                                  float* __restrict__ dst) {
  int idx = blockIdx.x * 256 + threadIdx.x;
  if (idx < BATCH * NVAL_ * 3) dst[idx] = src[idx];
}

__global__ __launch_bounds__(256) void loss_partial_k(
    const float* __restrict__ pred, const float* __restrict__ tgt,
    float* __restrict__ part) {
  __shared__ float red[256];
  int tid = threadIdx.x;
  int e = blockIdx.x * 256 + tid;
  float d = pred[e] - tgt[e];
  red[tid] = d * d;
  __syncthreads();
  for (int off = 128; off > 0; off >>= 1) {
    if (tid < off) red[tid] += red[tid + off];
    __syncthreads();
  }
  if (tid == 0) part[blockIdx.x] = red[0];
}

__global__ __launch_bounds__(128) void loss_final_k(const float* __restrict__ part,
                                                    float* __restrict__ out) {
  __shared__ float red[128];
  int tid = threadIdx.x;
  red[tid] = (tid < 96) ? part[tid] : 0.0f;
  __syncthreads();
  for (int off = 64; off > 0; off >>= 1) {
    if (tid < off) red[tid] += red[tid + off];
    __syncthreads();
  }
  if (tid == 0) out[0] = red[0] * (1.0f / (BATCH * NVAL_ * 3));
}

// ------------------------------- host side ---------------------------------

extern "C" void kernel_launch(void* const* d_in, const int* in_sizes, int n_in,
                              void* d_out, int out_size, void* d_ws, size_t ws_size,
                              hipStream_t stream) {
  (void)in_sizes; (void)n_in; (void)out_size; (void)ws_size;
  // Inputs in setup_inputs() dict (insertion) order, depth-first:
  const float* im8       = (const float*)d_in[0];
  const float* im64_tgt  = (const float*)d_in[1];
  const float* from_rgb_w = (const float*)d_in[2];
  const float* from_rgb_b = (const float*)d_in[3];
  const float* to_rgb_w  = (const float*)d_in[4];
  const float* to_rgb_b  = (const float*)d_in[5];
  const float* s0        = (const float*)d_in[6];
  const float* s1        = (const float*)d_in[7];
  const float* suffix    = (const float*)d_in[8];
  struct Blk {
    const float *ln_g, *ln_b, *in_proj, *conv_w, *conv_b, *dt_bias, *A_log, *D,
        *norm_w, *out_proj;
  };
  Blk blk[4];
  for (int i = 0; i < 4; ++i) {
    int o = 9 + i * 10;
    blk[i].ln_g    = (const float*)d_in[o + 0];
    blk[i].ln_b    = (const float*)d_in[o + 1];
    blk[i].in_proj = (const float*)d_in[o + 2];
    blk[i].conv_w  = (const float*)d_in[o + 3];
    blk[i].conv_b  = (const float*)d_in[o + 4];
    blk[i].dt_bias = (const float*)d_in[o + 5];
    blk[i].A_log   = (const float*)d_in[o + 6];
    blk[i].D       = (const float*)d_in[o + 7];
    blk[i].norm_w  = (const float*)d_in[o + 8];
    blk[i].out_proj = (const float*)d_in[o + 9];
  }

  // Workspace carve-out (~397 MB)
  char* p = (char*)d_ws;
  auto alloc = [&](size_t bytes) -> void* {
    void* r = (void*)p;
    p += (bytes + 255) & ~(size_t)255;
    return r;
  };
  bf16* wIn[4];  for (int i = 0; i < 4; ++i) wIn[i]  = (bf16*)alloc((size_t)D_MODEL * D_IN_PROJ * 2);
  bf16* wOut[4]; for (int i = 0; i < 4; ++i) wOut[i] = (bf16*)alloc((size_t)D_INNER * D_MODEL * 2);
  float* X    = (float*)alloc((size_t)NTOK * D_MODEL * 4);
  float* Xs   = (float*)alloc((size_t)NTOK * D_MODEL * 4);
  float* accA = (float*)alloc((size_t)NTOK * D_MODEL * 4);
  float* accB = (float*)alloc((size_t)NTOK * D_MODEL * 4);
  bf16* xln_bf = (bf16*)alloc((size_t)NTOK * D_MODEL * 2);
  float* zxbcdt = (float*)alloc((size_t)NTOK * D_IN_PROJ * 4);
  bf16* xs_bf = (bf16*)alloc((size_t)BATCH * LPAD * D_INNER * 2);
  bf16* Bc_bf = (bf16*)alloc((size_t)BATCH * LPAD * 64 * 2);
  bf16* Cc_bf = (bf16*)alloc((size_t)BATCH * LPAD * 64 * 2);
  float* dtp  = (float*)alloc((size_t)BATCH * LPAD * NHEADS * 4);
  float* csp  = (float*)alloc((size_t)BATCH * LPAD * NHEADS * 4);
  float* scores  = (float*)alloc((size_t)NBC * 65536 * 4);
  float* Sbuf    = (float*)alloc((size_t)NBC * NHEADS * 4096 * 4);
  float* prevbuf = (float*)alloc((size_t)NBC * NHEADS * 4096 * 4);
  float* ybuf    = (float*)alloc((size_t)BATCH * LPAD * D_INNER * 4);
  bf16* ygate_bf = (bf16*)alloc((size_t)NTOK * D_INNER * 2);
  float* im64buf = (float*)alloc((size_t)BATCH * NVAL_ * 3 * 4);
  float* losspart = (float*)alloc(1024 * 4);

  // Convert weights to bf16
  for (int i = 0; i < 4; ++i) {
    int n1 = D_MODEL * D_IN_PROJ;
    cvt_bf16_k<<<(n1 + 255) / 256, 256, 0, stream>>>(blk[i].in_proj, wIn[i], n1);
    int n2 = D_INNER * D_MODEL;
    cvt_bf16_k<<<(n2 + 255) / 256, 256, 0, stream>>>(blk[i].out_proj, wOut[i], n2);
  }

  auto run_mamba = [&](const float* xin, const Blk& bb, const bf16* wIn_,
                       const bf16* wOut_, float* out) {
    layernorm_bf16_k<<<NTOK, 256, 0, stream>>>(xin, bb.ln_g, bb.ln_b, xln_bf);
    dim3 g1((D_IN_PROJ + GBN - 1) / GBN, (NTOK + GBM - 1) / GBM);
    gemm_bf16_rr<<<g1, 256, 0, stream>>>(xln_bf, wIn_, zxbcdt, nullptr,
                                         NTOK, D_IN_PROJ, D_MODEL);
    dtsoft_k<<<(BATCH * LPAD * NHEADS + 255) / 256, 256, 0, stream>>>(
        zxbcdt, bb.dt_bias, dtp);
    convsilu_k<<<(BATCH * LPAD * D_XBC + 255) / 256, 256, 0, stream>>>(
        zxbcdt, bb.conv_w, bb.conv_b, xs_bf, Bc_bf, Cc_bf);
    cs_k<<<(NBC * NHEADS + 255) / 256, 256, 0, stream>>>(dtp, bb.A_log, csp);
    ssd_scores_k<<<dim3(2, 2, NBC), 256, 0, stream>>>(Cc_bf, Bc_bf, scores);
    ssd_yintra_k<<<dim3(NHEADS, NBC), 256, 0, stream>>>(scores, xs_bf, csp, dtp, ybuf);
    ssd_S_k<<<dim3(NHEADS, NBC), 256, 0, stream>>>(Bc_bf, xs_bf, csp, dtp, Sbuf);
    ssd_scan_k<<<BATCH * NHEADS, 256, 0, stream>>>(Sbuf, csp, prevbuf);
    ssd_yinter_k<<<dim3(NHEADS, NBC), 256, 0, stream>>>(Cc_bf, prevbuf, csp, xs_bf,
                                                        bb.D, ybuf);
    gaterms_k<<<NTOK, 256, 0, stream>>>(ybuf, zxbcdt, bb.norm_w, ygate_bf);
    dim3 g2((D_MODEL + GBN - 1) / GBN, (NTOK + GBM - 1) / GBM);
    gemm_bf16_rr<<<g2, 256, 0, stream>>>(ygate_bf, wOut_, out, xin,
                                         NTOK, D_MODEL, D_INNER);
  };

  resize_k<<<(BATCH * NVAL_ * 3 + 255) / 256, 256, 0, stream>>>(im8, im64buf);

  int nxe = NTOK * D_MODEL;
  for (int step = 0; step < 3; ++step) {
    embed_k<<<(nxe + 255) / 256, 256, 0, stream>>>(im8, im64buf, from_rgb_w,
                                                   from_rgb_b, s0, s1, suffix, X);
    for (int li = 0; li < 2; ++li) {
      run_mamba(X, blk[2 * li], wIn[2 * li], wOut[2 * li], accA);        // lr
      perm_k<<<(nxe + 255) / 256, 256, 0, stream>>>(X, Xs);
      run_mamba(Xs, blk[2 * li + 1], wIn[2 * li + 1], wOut[2 * li + 1], accB);
      combine_k<<<(nxe + 255) / 256, 256, 0, stream>>>(accA, accB, X);   // lr+du
    }
    torgb_k<<<BATCH * NVAL_, 256, 0, stream>>>(X, to_rgb_w, to_rgb_b, im64buf);
  }

  predcopy_k<<<(BATCH * NVAL_ * 3 + 255) / 256, 256, 0, stream>>>(im64buf,
                                                                  (float*)d_out);
  loss_partial_k<<<96, 256, 0, stream>>>(im64buf, im64_tgt, losspart);
  loss_final_k<<<1, 128, 0, stream>>>(losspart, (float*)d_out + BATCH * NVAL_ * 3);
}
